// MSPAMamba_9749575762045
// MI455X (gfx1250) — matrix operand order repaired and checked by reading, not compiled, not run on hardware
//
#include <hip/hip_runtime.h>
#include <hip/hip_bf16.h>

typedef __attribute__((ext_vector_type(16))) _Float16 v16h;
typedef __attribute__((ext_vector_type(8)))  _Float16 v8h;
typedef __attribute__((ext_vector_type(8)))  float    v8f;

#define BB 4
#define CCH 512
#define HW 1024
#define NROWS 4096      // BB*HW
#define NHEADS 16
#define DHEAD 32
#define CH4N 128
#define N1 256          // pooled positions (16x16)
#define DI_ 1024
#define DS_ 16
#define DTR_ 32
#define DIM2 2048       // 2*DI

enum { EPI_GB=1, EPI_RELU6=2, EPI_SOFTPLUS=4, EPI_SCALE=8, ACC_IN=16, ACC_OUT=32, EPI_MUL=64 };

enum {
  IN_X=0, IN_NORM1_G, IN_NORM1_B, IN_MSC_W01, IN_BN01_G, IN_BN01_B,
  IN_MSC_W02, IN_BN02_G, IN_BN02_B, IN_MSC_W03, IN_BN03_G, IN_BN03_B,
  IN_MSC_W12, IN_BN12_G, IN_BN12_B, IN_MSC_W13, IN_BN13_G, IN_BN13_B,
  IN_MSC_W22, IN_BN22_G, IN_BN22_B, IN_MSC_W23, IN_BN23_G, IN_BN23_B,
  IN_MSC_W3, IN_BN3_G, IN_BN3_B, IN_KV_W, IN_FC1_W, IN_FC1_B,
  IN_FC2_W, IN_FC2_B, IN_LN_G, IN_LN_B, IN_IN_W, IN_CONV_W, IN_CONV_B,
  IN_XPROJ_W, IN_DT_W, IN_DT_B, IN_A_LOG, IN_DP, IN_OUT_W,
  IN_FUSION_W, IN_FBN_G, IN_FBN_B, IN_MFC1_W, IN_MBN1_G, IN_MBN1_B,
  IN_MC1_W, IN_MBNC1_G, IN_MBNC1_B, IN_MC2_W, IN_MBNC2_G, IN_MBNC2_B,
  IN_MFC2_W, IN_MBNF2_G, IN_MBNF2_B, IN_ALPHA
};

// ---------------------------------------------------------------------------
// Generic batched WMMA GEMM:  Out[n][m] = sum_k A[n][k] * Bw[m][k]
// A: f16 activation rows (lda stride). Bw: f16 "weight" rows (ldb stride).
// z-batch decomposed as (b = z/nH, h = z%nH) with per-b / per-h strides.
// 4 waves/block. Each wave: 32 rows (2 A-frags) x (CF*16) cols.
// K loop is unrolled x2 with STATICALLY-indexed double buffers so all
// fragments stay in VGPRs (no scratch/LDS spill) and loads for step i+1
// are in flight while the WMMAs of step i execute.
// ---------------------------------------------------------------------------
template<int CF>
__global__ __launch_bounds__(128)
void wmma_gemm(const _Float16* __restrict__ A, int lda, long sAb, long sAh,
               const _Float16* __restrict__ Bw, int ldb, long sBb, long sBh,
               float* __restrict__ Cf, _Float16* __restrict__ Ch,
               int ldc, long sCb, long sCh,
               const float* __restrict__ mulsrc,
               int K, int nH,
               const float* __restrict__ eg, const float* __restrict__ ebv,
               float scale, int mode)
{
  const int lane = threadIdx.x & 31;
  const int wave = threadIdx.x >> 5;
  const int sub  = lane >> 4;     // 0: lanes 0-15, 1: lanes 16-31
  const int l15  = lane & 15;
  const int z  = blockIdx.z;
  const int bb = z / nH, hh = z % nH;
  const long aoff = (long)bb*sAb + (long)hh*sAh;
  const long boff = (long)bb*sBb + (long)hh*sBh;
  const long coff = (long)bb*sCb + (long)hh*sCh;
  const int row0 = ((int)blockIdx.y*4 + wave)*32;
  const int col0 = (int)blockIdx.x * (CF*16);

  v8f zero8 = {0.f,0.f,0.f,0.f,0.f,0.f,0.f,0.f};
  v8f acc[2][CF];
  #pragma unroll
  for (int rf = 0; rf < 2; ++rf)
    #pragma unroll
    for (int j = 0; j < CF; ++j) acc[rf][j] = zero8;

  const _Float16* arow0 = A + aoff + (long)(row0 + l15)*lda;
  const _Float16* arow1 = arow0 + (long)16*lda;
  const _Float16* bbase = Bw + boff + (long)(col0 + l15)*ldb;

  union H16 { v16h v; v8h p[2]; };
  H16 a0[2], a1[2];     // [buf] x rowfrag fragments
  H16 b0[2][CF];        // [buf][colfrag]

  // A 16x32 f16 fragment: lane holds row (l15), K = k0 + sub*16 + [0..15] contiguous
  // B 32x16 fragment: lane holds column (l15), K = k0 + sub*16 + [0..15] contiguous
  #define LOAD_FRAGS(BUF, K0)                                                  \
    do {                                                                       \
      a0[BUF].p[0] = *(const v8h*)(arow0 + (K0) + sub*16);                     \
      a0[BUF].p[1] = *(const v8h*)(arow0 + (K0) + sub*16 + 8);                 \
      a1[BUF].p[0] = *(const v8h*)(arow1 + (K0) + sub*16);                     \
      a1[BUF].p[1] = *(const v8h*)(arow1 + (K0) + sub*16 + 8);                 \
      _Pragma("unroll")                                                        \
      for (int j = 0; j < CF; ++j)                                             \
        b0[BUF][j].v = *(const v16h*)(bbase + (long)j*16*ldb + (K0) + sub*16); \
    } while (0)

  #define MMA_STEP(BUF)                                                        \
    do {                                                                       \
      _Pragma("unroll")                                                        \
      for (int j = 0; j < CF; ++j) {                                           \
        acc[0][j] = __builtin_amdgcn_wmma_f32_16x16x32_f16(                    \
                      false, a0[BUF].v, false, b0[BUF][j].v, (short)0,         \
                      acc[0][j], false, false);                                \
        acc[1][j] = __builtin_amdgcn_wmma_f32_16x16x32_f16(                    \
                      false, a1[BUF].v, false, b0[BUF][j].v, (short)0,         \
                      acc[1][j], false, false);                                \
      }                                                                        \
    } while (0)

  const int nsteps = K >> 5;
  LOAD_FRAGS(0, 0);
  int k0 = 0;
  for (int s = 0; s + 2 <= nsteps; s += 2) {
    LOAD_FRAGS(1, k0 + 32);
    MMA_STEP(0);
    if (s + 2 < nsteps) LOAD_FRAGS(0, k0 + 64);
    MMA_STEP(1);
    k0 += 64;
  }
  if (nsteps & 1) MMA_STEP(0);

  #undef LOAD_FRAGS
  #undef MMA_STEP

  #pragma unroll
  for (int rf = 0; rf < 2; ++rf) {
    #pragma unroll
    for (int j = 0; j < CF; ++j) {
      const int col = col0 + j*16 + l15;
      const float g  = (mode & EPI_GB) ? (eg  ? eg[col]  : 1.0f) : 1.0f;
      const float bv = (mode & EPI_GB) ? (ebv ? ebv[col] : 0.0f) : 0.0f;
      union { v8f v; float f[8]; } u; u.v = acc[rf][j];
      #pragma unroll
      for (int r = 0; r < 8; ++r) {
        const int row = row0 + rf*16 + r + sub*8; // D: vgpr r -> M=r / M=r+8
        const long idx = coff + (long)row*ldc + col;
        float y = u.f[r];
        if (mode & ACC_IN)      y += Cf[idx];
        if (mode & EPI_GB)      y = y*g + bv;
        if (mode & EPI_RELU6)   y = fminf(fmaxf(y, 0.f), 6.f);
        if (mode & EPI_SOFTPLUS)y = (y > 20.f) ? y : log1pf(__expf(y));
        if (mode & EPI_SCALE)   y *= scale;
        if (mode & EPI_MUL)     y *= mulsrc[idx];
        if (mode & ACC_OUT)     y += Cf[idx];
        if (Cf) Cf[idx] = y;
        if (Ch) Ch[idx] = (_Float16)y;
      }
    }
  }
}

// ---------------------------------------------------------------------------
__global__ void f32_to_f16(const float* __restrict__ s, _Float16* __restrict__ d, int n) {
  int i = blockIdx.x*256 + threadIdx.x;
  if (i < n) d[i] = (_Float16)s[i];
}

// x NCHW f32 -> pixel-major x_pm f32, xn = bn(x,norm1) f32 + f16
__global__ void input_norm(const float* __restrict__ x, const float* __restrict__ g,
                           const float* __restrict__ bt, float* __restrict__ xpm,
                           float* __restrict__ xnf, _Float16* __restrict__ xnh) {
  long i = (long)blockIdx.x*256 + threadIdx.x;
  if (i >= (long)NROWS*CCH) return;
  int c = (int)(i % CCH); long rp = i / CCH; int p = (int)(rp % HW); int b = (int)(rp / HW);
  float v = x[((long)(b*CCH + c))*HW + p];
  xpm[i] = v;
  float n = v*g[c] + bt[c];
  xnf[i] = n; xnh[i] = (_Float16)n;
}

// depthwise kxk dilated conv on pixel-major f16, bn + relu6 (+ optional add)
__global__ void dwconv2d(const _Float16* __restrict__ in, _Float16* __restrict__ out,
                         const float* __restrict__ w, const float* __restrict__ g,
                         const float* __restrict__ bt, int Cc, int ks, int dil,
                         const _Float16* __restrict__ addsrc) {
  long i = (long)blockIdx.x*256 + threadIdx.x;
  if (i >= (long)NROWS*Cc) return;
  int c = (int)(i % Cc); long rp = i / Cc; int p = (int)(rp % HW); int b = (int)(rp / HW);
  int py = p >> 5, px = p & 31, rad = ks >> 1;
  float acc = 0.f;
  for (int ky = 0; ky < ks; ++ky) {
    int iy = py + (ky - rad)*dil;
    if ((unsigned)iy >= 32u) continue;
    for (int kx = 0; kx < ks; ++kx) {
      int ix = px + (kx - rad)*dil;
      if ((unsigned)ix >= 32u) continue;
      acc += (float)in[((long)(b*HW + iy*32 + ix))*Cc + c] * w[(c*ks + ky)*ks + kx];
    }
  }
  float y = acc*g[c] + bt[c];
  y = fminf(fmaxf(y, 0.f), 6.f);
  if (addsrc) y += (float)addsrc[i];
  out[i] = (_Float16)y;
}

// mean over 2x2 blocks -> (B,256,512) f16 pixel-major
__global__ void pool_mean(const float* __restrict__ a, _Float16* __restrict__ out) {
  long i = (long)blockIdx.x*256 + threadIdx.x;
  if (i >= (long)BB*N1*CCH) return;
  int c = (int)(i % CCH); long rp = i / CCH; int p = (int)(rp % N1); int b = (int)(rp / N1);
  int p1 = p >> 4, p2 = p & 15;
  float s = 0.f;
  for (int ii = 0; ii < 2; ++ii)
    for (int jj = 0; jj < 2; ++jj)
      s += a[((long)(b*HW + (p1*2+ii)*32 + (p2*2+jj)))*CCH + c];
  out[i] = (_Float16)(s*0.25f);
}

// Vt[b][h][dd][n1] = kvp[b][n1][512 + h*32 + dd]
__global__ void build_vt(const _Float16* __restrict__ kvp, _Float16* __restrict__ vt) {
  long i = (long)blockIdx.x*256 + threadIdx.x;
  if (i >= (long)BB*NHEADS*DHEAD*N1) return;
  int n1 = (int)(i & 255); long r = i >> 8;
  int dd = (int)(r & 31); r >>= 5;
  int h  = (int)(r & 15); int b = (int)(r >> 4);
  vt[i] = kvp[((long)(b*N1 + n1))*1024 + 512 + h*32 + dd];
}

// softmax over 256-wide rows of f16, in place. one wave per row.
__global__ void softmax_rows(_Float16* __restrict__ p, long nrowstotal) {
  int lane = threadIdx.x & 31; int wv = threadIdx.x >> 5;
  long row = (long)blockIdx.x*8 + wv;
  if (row >= nrowstotal) return;
  _Float16* pr = p + row*256;
  union { v8h v; _Float16 h[8]; } d;
  d.v = *(const v8h*)(pr + lane*8);
  float v[8]; float m = -1e30f;
  #pragma unroll
  for (int i = 0; i < 8; ++i) { v[i] = (float)d.h[i]; m = fmaxf(m, v[i]); }
  for (int o = 16; o > 0; o >>= 1) m = fmaxf(m, __shfl_xor(m, o, 32));
  float s = 0.f;
  #pragma unroll
  for (int i = 0; i < 8; ++i) { v[i] = __expf(v[i] - m); s += v[i]; }
  for (int o = 16; o > 0; o >>= 1) s += __shfl_xor(s, o, 32);
  float inv = 1.f/s;
  #pragma unroll
  for (int i = 0; i < 8; ++i) d.h[i] = (_Float16)(v[i]*inv);
  *(v8h*)(pr + lane*8) = d.v;
}

// per-(b,c) mean over HW pixels of xn
__global__ void colmean(const float* __restrict__ x, float* __restrict__ out) {
  int b = blockIdx.x; int c = threadIdx.x;  // 512 threads
  const float* p = x + (long)b*HW*CCH + c;
  float s = 0.f;
  for (int r = 0; r < HW; ++r) s += p[(long)r*CCH];
  out[b*CCH + c] = s * (1.0f/HW);
}

// channel attention: fc1 (C->128) relu6, fc2 (128->C) sigmoid
__global__ void chattn_fc(const float* __restrict__ camean,
                          const float* __restrict__ fc1w, const float* __restrict__ fc1b,
                          const float* __restrict__ fc2w, const float* __restrict__ fc2b,
                          float* __restrict__ ca) {
  __shared__ float s1[CH4N];
  int b = blockIdx.x; int t = threadIdx.x;  // 128 threads
  const float* cm = camean + b*CCH;
  float acc = fc1b[t];
  for (int c = 0; c < CCH; ++c) acc += cm[c]*fc1w[t*CCH + c];
  s1[t] = fminf(fmaxf(acc, 0.f), 6.f);
  __syncthreads();
  for (int o = t; o < CCH; o += CH4N) {
    float a2 = fc2b[o];
    for (int j = 0; j < CH4N; ++j) a2 += s1[j]*fc2w[o*CH4N + j];
    ca[b*CCH + o] = 1.f/(1.f + __expf(-a2));
  }
}

// attn_out += ca[b][c]*xn ; also emit f16
__global__ void attn_add(float* __restrict__ attn, _Float16* __restrict__ attnh,
                         const float* __restrict__ ca, const float* __restrict__ xn) {
  long i = (long)blockIdx.x*256 + threadIdx.x;
  if (i >= (long)NROWS*CCH) return;
  int c = (int)(i % CCH); int b = (int)(i / ((long)CCH*HW));
  float y = attn[i] + ca[b*CCH + c]*xn[i];
  attn[i] = y; attnh[i] = (_Float16)y;
}

// layernorm over C per pixel-row of x_pm -> f16
__global__ void layernorm_rows(const float* __restrict__ x, _Float16* __restrict__ out,
                               const float* __restrict__ g, const float* __restrict__ bt) {
  int lane = threadIdx.x & 31; int wv = threadIdx.x >> 5;
  long row = (long)blockIdx.x*8 + wv;
  if (row >= NROWS) return;
  const float* xr = x + row*CCH;
  float v[16]; float s = 0.f;
  #pragma unroll
  for (int i = 0; i < 16; ++i) { v[i] = xr[lane*16 + i]; s += v[i]; }
  for (int o = 16; o > 0; o >>= 1) s += __shfl_xor(s, o, 32);
  float mu = s*(1.f/CCH);
  float q = 0.f;
  #pragma unroll
  for (int i = 0; i < 16; ++i) { float d = v[i]-mu; q += d*d; }
  for (int o = 16; o > 0; o >>= 1) q += __shfl_xor(q, o, 32);
  float rstd = rsqrtf(q*(1.f/CCH) + 1e-5f);
  #pragma unroll
  for (int i = 0; i < 16; ++i) {
    int c = lane*16 + i;
    out[row*CCH + c] = (_Float16)((v[i]-mu)*rstd*g[c] + bt[c]);
  }
}

// causal depthwise conv1d (k=3) + silu on xi = xz[:, :DI]
__global__ void dwconv1d_silu(const float* __restrict__ xz, const float* __restrict__ w,
                              const float* __restrict__ bi, float* __restrict__ xc,
                              _Float16* __restrict__ xch) {
  long i = (long)blockIdx.x*256 + threadIdx.x;
  if (i >= (long)NROWS*DI_) return;
  int ch = (int)(i % DI_); long rl = i / DI_; int l = (int)(rl % HW); int b = (int)(rl / HW);
  const float* xi = xz + (long)(b*HW)*DIM2 + ch;
  float acc = bi[ch];
  #pragma unroll
  for (int t = 0; t < 3; ++t) {
    int ll = l - 2 + t;
    if (ll >= 0) acc += xi[(long)ll*DIM2]*w[ch*3 + t];
  }
  float sv = acc/(1.f + __expf(-acc));
  xc[i] = sv; xch[i] = (_Float16)sv;
}

// selective scan: one thread per (b, DI-channel), 16-state
__global__ void mamba_scan(const float* __restrict__ delta, const float* __restrict__ xc,
                           const float* __restrict__ dbl, const float* __restrict__ A_log,
                           const float* __restrict__ Dp, float* __restrict__ y) {
  int ch = blockIdx.x*blockDim.x + threadIdx.x;
  int b  = blockIdx.y;
  if (ch >= DI_) return;
  float A[DS_], h[DS_];
  #pragma unroll
  for (int s = 0; s < DS_; ++s) { A[s] = -__expf(A_log[ch*DS_ + s]); h[s] = 0.f; }
  float Dv = Dp[ch];
  const float* dlt = delta + (long)b*HW*DI_ + ch;
  const float* xcp = xc    + (long)b*HW*DI_ + ch;
  const float* dbp = dbl   + (long)b*HW*64;
  float* yp = y + (long)b*HW*DI_ + ch;
  for (int l = 0; l < HW; ++l) {
    float dl = dlt[(long)l*DI_];
    float xv = xcp[(long)l*DI_];
    const float* bc = dbp + l*64;
    float acc = 0.f;
    #pragma unroll
    for (int s = 0; s < DS_; ++s) {
      float dA = __expf(dl*A[s]);
      h[s] = dA*h[s] + dl*bc[32 + s]*xv;
      acc += h[s]*bc[48 + s];
    }
    yp[(long)l*DI_] = acc + xv*Dv;
  }
}

// y * silu(z) -> f16
__global__ void gate_silu(const float* __restrict__ y, const float* __restrict__ xz,
                          _Float16* __restrict__ yh) {
  long i = (long)blockIdx.x*256 + threadIdx.x;
  if (i >= (long)NROWS*DI_) return;
  int ch = (int)(i % DI_); long r = i / DI_;
  float zv = xz[r*DIM2 + DI_ + ch];
  float sv = zv/(1.f + __expf(-zv));
  yh[i] = (_Float16)(y[i]*sv);
}

// x1 = x + a*attn + (1-a)*fused
__global__ void combine_x1(const float* __restrict__ x, const float* __restrict__ attn,
                           const float* __restrict__ fused, const float* __restrict__ alpha,
                           float* __restrict__ x1, _Float16* __restrict__ x1h) {
  long i = (long)blockIdx.x*256 + threadIdx.x;
  if (i >= (long)NROWS*CCH) return;
  float a = alpha[0];
  float v = x[i] + a*attn[i] + (1.f - a)*fused[i];
  x1[i] = v; x1h[i] = (_Float16)v;
}

// out NCHW = x1 + t (pixel-major -> NCHW)
__global__ void final_out(const float* __restrict__ x1, const float* __restrict__ t,
                          float* __restrict__ out) {
  long i = (long)blockIdx.x*256 + threadIdx.x;
  if (i >= (long)BB*CCH*HW) return;
  int p = (int)(i % HW); long rc = i / HW; int c = (int)(rc % CCH); int b = (int)(rc / CCH);
  long pm = ((long)(b*HW + p))*CCH + c;
  out[i] = x1[pm] + t[pm];
}

// ---------------------------------------------------------------------------
extern "C" void kernel_launch(void* const* d_in, const int* in_sizes, int n_in,
                              void* d_out, int out_size, void* d_ws, size_t ws_size,
                              hipStream_t stream) {
  (void)in_sizes; (void)n_in; (void)out_size; (void)ws_size;
  const float* F[64];
  for (int i = 0; i < 59 && i < n_in; ++i) F[i] = (const float*)d_in[i];

  char* base = (char*)d_ws; size_t cur = 0;
  auto alloc = [&](size_t bytes) -> char* {
    char* p = base + cur; cur += (bytes + 255) & ~(size_t)255; return p;
  };
  auto allocF = [&](size_t n) -> float*    { return (float*)alloc(n*4); };
  auto allocH = [&](size_t n) -> _Float16* { return (_Float16*)alloc(n*2); };

  // f16 weights
  _Float16* w01h   = allocH(128*512);
  _Float16* w03h   = allocH(512*128);
  _Float16* w13h   = allocH(512*128);
  _Float16* w23h   = allocH(512*128);
  _Float16* w3h    = allocH(512*512);
  _Float16* kvwh   = allocH(1024*512);
  _Float16* inwh   = allocH(2048*512);
  _Float16* xprojh = allocH(64*1024);
  _Float16* dtwh   = allocH(1024*32);
  _Float16* outwh  = allocH(512*1024);
  _Float16* fuswh  = allocH(512*1024);
  _Float16* mfc1h  = allocH(128*512);
  _Float16* mfc2h  = allocH(512*128);

  // activations
  float*    x_pm   = allocF((size_t)NROWS*CCH);
  float*    xn_f   = allocF((size_t)NROWS*CCH);
  _Float16* xn_h   = allocH((size_t)NROWS*CCH);
  _Float16* a01_h  = allocH((size_t)NROWS*CH4N);
  _Float16* d0h    = allocH((size_t)NROWS*CH4N);
  _Float16* d1h    = allocH((size_t)NROWS*CH4N);
  _Float16* d2h    = allocH((size_t)NROWS*CH4N);
  float*    gsum   = allocF((size_t)NROWS*CCH);
  _Float16* gsum_h = allocH((size_t)NROWS*CCH);
  float*    attnp  = allocF((size_t)NROWS*CCH);
  _Float16* pooled = allocH((size_t)BB*N1*CCH);
  _Float16* kvp_h  = allocH((size_t)BB*N1*1024);
  _Float16* vt_h   = allocH((size_t)BB*NHEADS*DHEAD*N1);
  _Float16* probs  = allocH((size_t)BB*NHEADS*HW*N1);
  float*    attn_o = allocF((size_t)NROWS*CCH);
  _Float16* attn_h = allocH((size_t)NROWS*CCH);
  float*    camean = allocF((size_t)BB*CCH);
  float*    ca     = allocF((size_t)BB*CCH);
  _Float16* ln_h   = allocH((size_t)NROWS*CCH);
  float*    xz     = allocF((size_t)NROWS*DIM2);
  float*    xc     = allocF((size_t)NROWS*DI_);
  _Float16* xc_h   = allocH((size_t)NROWS*DI_);
  float*    dbl    = allocF((size_t)NROWS*64);
  _Float16* dbl_h  = allocH((size_t)NROWS*64);
  float*    delta  = allocF((size_t)NROWS*DI_);
  float*    ybuf   = allocF((size_t)NROWS*DI_);
  _Float16* yh     = allocH((size_t)NROWS*DI_);
  _Float16* mo_h   = allocH((size_t)NROWS*CCH);
  float*    fsum   = allocF((size_t)NROWS*CCH);
  float*    x1     = allocF((size_t)NROWS*CCH);
  _Float16* x1_h   = allocH((size_t)NROWS*CCH);
  _Float16* t0_h   = allocH((size_t)NROWS*CH4N);
  _Float16* t1_h   = allocH((size_t)NROWS*CH4N);
  _Float16* ts_h   = allocH((size_t)NROWS*CH4N);
  float*    tfin   = allocF((size_t)NROWS*CCH);

  auto cvt = [&](const float* s, _Float16* d, int n) {
    f32_to_f16<<<(n + 255)/256, 256, 0, stream>>>(s, d, n);
  };
  cvt(F[IN_MSC_W01], w01h,  128*512);
  cvt(F[IN_MSC_W03], w03h,  512*128);
  cvt(F[IN_MSC_W13], w13h,  512*128);
  cvt(F[IN_MSC_W23], w23h,  512*128);
  cvt(F[IN_MSC_W3],  w3h,   512*512);
  cvt(F[IN_KV_W],    kvwh,  1024*512);
  cvt(F[IN_IN_W],    inwh,  2048*512);
  cvt(F[IN_XPROJ_W], xprojh, 64*1024);
  cvt(F[IN_DT_W],    dtwh,  1024*32);
  cvt(F[IN_OUT_W],   outwh, 512*1024);
  cvt(F[IN_FUSION_W],fuswh, 512*1024);
  cvt(F[IN_MFC1_W],  mfc1h, 128*512);
  cvt(F[IN_MFC2_W],  mfc2h, 512*128);

  auto gemm = [&](const _Float16* A, int lda, long sAb, long sAh,
                  const _Float16* Bw, int ldb, long sBb, long sBh,
                  float* Cf, _Float16* Ch, int ldc, long sCb, long sCh,
                  const float* mul, int M, int K, int nH, int nbatch, int Nrows,
                  const float* eg, const float* eb, float scale, int mode, int cfr) {
    dim3 grid((unsigned)(M/(16*cfr)), (unsigned)(Nrows/128), (unsigned)nbatch);
    if (cfr == 2)
      wmma_gemm<2><<<grid, 128, 0, stream>>>(A, lda, sAb, sAh, Bw, ldb, sBb, sBh,
                                             Cf, Ch, ldc, sCb, sCh, mul, K, nH,
                                             eg, eb, scale, mode);
    else
      wmma_gemm<4><<<grid, 128, 0, stream>>>(A, lda, sAb, sAh, Bw, ldb, sBb, sBh,
                                             Cf, Ch, ldc, sCb, sCh, mul, K, nH,
                                             eg, eb, scale, mode);
  };

  const long NE = (long)NROWS*CCH;
  // input transpose + norm1
  input_norm<<<(unsigned)((NE + 255)/256), 256, 0, stream>>>(
      F[IN_X], F[IN_NORM1_G], F[IN_NORM1_B], x_pm, xn_f, xn_h);

  // ---- MSC ----
  gemm(xn_h, 512, 0,0, w01h, 512, 0,0, nullptr, a01_h, CH4N, 0,0, nullptr,
       128, 512, 1, 1, NROWS, F[IN_BN01_G], F[IN_BN01_B], 1.f, EPI_GB|EPI_RELU6, 4);

  long ND = (long)NROWS*CH4N;
  dwconv2d<<<(unsigned)((ND+255)/256),256,0,stream>>>(a01_h, d0h, F[IN_MSC_W02], F[IN_BN02_G], F[IN_BN02_B], CH4N, 3, 3, nullptr);
  dwconv2d<<<(unsigned)((ND+255)/256),256,0,stream>>>(a01_h, d1h, F[IN_MSC_W12], F[IN_BN12_G], F[IN_BN12_B], CH4N, 3, 5, nullptr);
  dwconv2d<<<(unsigned)((ND+255)/256),256,0,stream>>>(a01_h, d2h, F[IN_MSC_W22], F[IN_BN22_G], F[IN_BN22_B], CH4N, 3, 7, nullptr);

  gemm(d0h, 128, 0,0, w03h, 128, 0,0, gsum, nullptr, CCH, 0,0, nullptr,
       512, 128, 1, 1, NROWS, F[IN_BN03_G], F[IN_BN03_B], 1.f, EPI_GB|EPI_RELU6, 4);
  gemm(d1h, 128, 0,0, w13h, 128, 0,0, gsum, nullptr, CCH, 0,0, nullptr,
       512, 128, 1, 1, NROWS, F[IN_BN13_G], F[IN_BN13_B], 1.f, EPI_GB|EPI_RELU6|ACC_OUT, 4);
  gemm(d2h, 128, 0,0, w23h, 128, 0,0, gsum, gsum_h, CCH, 0,0, nullptr,
       512, 128, 1, 1, NROWS, F[IN_BN23_G], F[IN_BN23_B], 1.f, EPI_GB|EPI_RELU6|ACC_OUT, 4);

  gemm(gsum_h, 512, 0,0, w3h, 512, 0,0, attnp, nullptr, CCH, 0,0, xn_f,
       512, 512, 1, 1, NROWS, F[IN_BN3_G], F[IN_BN3_B], 1.f, EPI_GB|EPI_RELU6|EPI_MUL, 4);

  long NP = (long)BB*N1*CCH;
  pool_mean<<<(unsigned)((NP+255)/256),256,0,stream>>>(attnp, pooled);

  // kv projection on pooled map (1024 rows)
  gemm(pooled, 512, 0,0, kvwh, 512, 0,0, nullptr, kvp_h, 1024, 0,0, nullptr,
       1024, 512, 1, 1, BB*N1, nullptr, nullptr, 1.f, 0, 4);

  long NV = (long)BB*NHEADS*DHEAD*N1;
  build_vt<<<(unsigned)((NV+255)/256),256,0,stream>>>(kvp_h, vt_h);

  // Q*K^T per (b,h): A = xn_h columns h*32.., B rows = keys (K=32)
  gemm(xn_h, 512, (long)HW*512, 32,
       kvp_h, 1024, (long)N1*1024, 32,
       nullptr, probs, N1, (long)NHEADS*HW*N1, (long)HW*N1, nullptr,
       N1, 32, NHEADS, BB*NHEADS, HW,
       nullptr, nullptr, 0.17677669529663687f, EPI_SCALE, 4);

  softmax_rows<<<(unsigned)(((long)BB*NHEADS*HW + 7)/8), 256, 0, stream>>>(
      probs, (long)BB*NHEADS*HW);

  // probs x V  (B = Vt rows = d-index, K = 256 keys)
  gemm(probs, N1, (long)NHEADS*HW*N1, (long)HW*N1,
       vt_h, N1, (long)NHEADS*DHEAD*N1, (long)DHEAD*N1,
       attn_o, nullptr, CCH, (long)HW*CCH, 32, nullptr,
       DHEAD, N1, NHEADS, BB*NHEADS, HW,
       nullptr, nullptr, 1.f, 0, 2);

  // channel attention
  colmean<<<BB, CCH, 0, stream>>>(xn_f, camean);
  chattn_fc<<<BB, CH4N, 0, stream>>>(camean, F[IN_FC1_W], F[IN_FC1_B],
                                     F[IN_FC2_W], F[IN_FC2_B], ca);
  attn_add<<<(unsigned)((NE+255)/256),256,0,stream>>>(attn_o, attn_h, ca, xn_f);

  // ---- Mamba ----
  layernorm_rows<<<(unsigned)((NROWS+7)/8), 256, 0, stream>>>(x_pm, ln_h, F[IN_LN_G], F[IN_LN_B]);

  gemm(ln_h, 512, 0,0, inwh, 512, 0,0, xz, nullptr, DIM2, 0,0, nullptr,
       DIM2, 512, 1, 1, NROWS, nullptr, nullptr, 1.f, 0, 4);

  long NC = (long)NROWS*DI_;
  dwconv1d_silu<<<(unsigned)((NC+255)/256),256,0,stream>>>(xz, F[IN_CONV_W], F[IN_CONV_B], xc, xc_h);

  gemm(xc_h, 1024, 0,0, xprojh, 1024, 0,0, dbl, dbl_h, 64, 0,0, nullptr,
       64, 1024, 1, 1, NROWS, nullptr, nullptr, 1.f, 0, 4);

  gemm(dbl_h, 64, 0,0, dtwh, 32, 0,0, delta, nullptr, DI_, 0,0, nullptr,
       DI_, 32, 1, 1, NROWS, nullptr, F[IN_DT_B], 1.f, EPI_GB|EPI_SOFTPLUS, 4);

  mamba_scan<<<dim3(DI_/256, BB), 256, 0, stream>>>(delta, xc, dbl, F[IN_A_LOG], F[IN_DP], ybuf);
  gate_silu<<<(unsigned)((NC+255)/256),256,0,stream>>>(ybuf, xz, yh);

  gemm(yh, 1024, 0,0, outwh, 1024, 0,0, nullptr, mo_h, CCH, 0,0, nullptr,
       CCH, 1024, 1, 1, NROWS, nullptr, nullptr, 1.f, 0, 4);

  // ---- fusion over concat(attn, mo) ----
  gemm(attn_h, 512, 0,0, fuswh, 1024, 0,0, fsum, nullptr, CCH, 0,0, nullptr,
       CCH, 512, 1, 1, NROWS, nullptr, nullptr, 1.f, 0, 4);
  gemm(mo_h, 512, 0,0, fuswh + 512, 1024, 0,0, fsum, nullptr, CCH, 0,0, nullptr,
       CCH, 512, 1, 1, NROWS, F[IN_FBN_G], F[IN_FBN_B], 1.f, EPI_GB|ACC_IN, 4);

  combine_x1<<<(unsigned)((NE+255)/256),256,0,stream>>>(x_pm, attn_o, fsum, F[IN_ALPHA], x1, x1_h);

  // ---- MFF ----
  gemm(x1_h, 512, 0,0, mfc1h, 512, 0,0, nullptr, t0_h, CH4N, 0,0, nullptr,
       128, 512, 1, 1, NROWS, F[IN_MBN1_G], F[IN_MBN1_B], 1.f, EPI_GB|EPI_RELU6, 4);
  dwconv2d<<<(unsigned)((ND+255)/256),256,0,stream>>>(t0_h, t1_h, F[IN_MC1_W], F[IN_MBNC1_G], F[IN_MBNC1_B], CH4N, 5, 1, nullptr);
  dwconv2d<<<(unsigned)((ND+255)/256),256,0,stream>>>(t0_h, ts_h, F[IN_MC2_W], F[IN_MBNC2_G], F[IN_MBNC2_B], CH4N, 3, 1, t1_h);
  gemm(ts_h, 128, 0,0, mfc2h, 128, 0,0, tfin, nullptr, CCH, 0,0, nullptr,
       CCH, 128, 1, 1, NROWS, F[IN_MBNF2_G], F[IN_MBNF2_B], 1.f, EPI_GB|EPI_RELU6, 4);

  long NO = (long)BB*CCH*HW;
  final_out<<<(unsigned)((NO+255)/256),256,0,stream>>>(x1, tfin, (float*)d_out);
}
